// LatticeGaussian_40793599377962
// MI455X (gfx1250) — matrix operand order repaired and checked by reading, not compiled, not run on hardware
//
#include <hip/hip_runtime.h>

typedef __attribute__((ext_vector_type(2)))  float    v2f;
typedef __attribute__((ext_vector_type(4)))  float    v4f;
typedef __attribute__((ext_vector_type(8)))  float    v8f;
typedef __attribute__((ext_vector_type(16))) _Float16 v16h;

#define NROW 8192
#define LCH  16
#define WPB  4            // waves per block (wave32)

#define LOG2E  1.44269504088896340736f
#define LOG2E2 2.88539008177792681472f   // 2*log2(e)

// ---------------- prep kernel 1: msq[i] = -log2(e) * |ref_i|^2 ----------------
__global__ __launch_bounds__(256) void k_msq(const float* __restrict__ ref,
                                             float* __restrict__ msq) {
  int i = blockIdx.x * 256 + threadIdx.x;
  if (i < NROW) {
    v4f r = *(const v4f*)(ref + (size_t)i * 4);
    msq[i] = -LOG2E * (r.x * r.x + r.y * r.y + r.z * r.z + r.w * r.w);
  }
}

// ---- prep kernel 2: pack (e^{-|r_j|^2} * U[j][:]) as f16 WMMA B-fragments.
// For j-tile t (32 rows), lane l, dword v:
//   upk[t*256 + l*8 + v] = pack( h(w(row)*U[row][c]), h(w(row+1)*U[row+1][c]) )
//   row = t*32 + 8*(l>=16) + 2v + 8*(v>=4), c = l&15   (16-bit 32x16 B layout)
__global__ __launch_bounds__(256) void k_upk(const float* __restrict__ U,
                                             const float* __restrict__ msq,
                                             unsigned int* __restrict__ upk) {
  int t = blockIdx.x * 256 + threadIdx.x;   // 0 .. 65535
  int tile = t >> 8;
  int r    = t & 255;
  int lane = r >> 3;
  int v    = r & 7;
  int c    = lane & 15;
  int hi   = lane >> 4;
  int row  = tile * 32 + 8 * hi + 2 * v + 8 * (v >> 2);
  float w0 = __builtin_amdgcn_exp2f(msq[row]);
  float w1 = __builtin_amdgcn_exp2f(msq[row + 1]);
  _Float16 h0 = (_Float16)(w0 * U[(size_t)row * LCH + c]);
  _Float16 h1 = (_Float16)(w1 * U[(size_t)(row + 1) * LCH + c]);
  unsigned int u0 = __builtin_bit_cast(unsigned short, h0);
  unsigned int u1 = __builtin_bit_cast(unsigned short, h1);
  upk[t] = u0 | (u1 << 16);
}

// ---------------- main fused kernel ----------------
// Per 32-j chunk:
//   GEMM1 (x2): S = ref_j(16x4) @ (2*log2e*ref_i)^T(4x16) + 0   [fp32 WMMA]
//   VALU:       G = exp2(S) -> f16   (S^T D-layout == GEMM2 A-frag element order)
//   GEMM2:      acc += G(16x32 f16) @ Upk(32x16 f16)
// Bias terms e^{-|rj|^2} / e^{-|ri|^2} are pre/post-folded (no clamp, no bias VALU).
__device__ __forceinline__ void gauss_chunk(const char* __restrict__ refB,
                                            const char* __restrict__ upkB,
                                            unsigned aOff, unsigned uOff,
                                            v2f bfrag, v8f& acc) {
  v2f a1 = *(const v2f*)(refB + aOff);          // rows jb..jb+15
  v2f a2 = *(const v2f*)(refB + aOff + 256);    // rows jb+16..jb+31
  v8f z = {};
  v8f s1 = __builtin_amdgcn_wmma_f32_16x16x4_f32(false, a1, false, bfrag,
                                                 (short)0, z, false, false);
  v8f s2 = __builtin_amdgcn_wmma_f32_16x16x4_f32(false, a2, false, bfrag,
                                                 (short)0, z, false, false);
  v16h ka;
#pragma unroll
  for (int e = 0; e < 8; ++e) {
    ka[e]     = (_Float16)__builtin_amdgcn_exp2f(s1[e]);
    ka[8 + e] = (_Float16)__builtin_amdgcn_exp2f(s2[e]);
  }
  v16h kb = *(const v16h*)(upkB + uOff);        // prepacked 32B/lane
  acc = __builtin_amdgcn_wmma_f32_16x16x32_f16(false, ka, false, kb,
                                               (short)0, acc, false, false);
}

__global__ __launch_bounds__(WPB * 32) void k_main(const float* __restrict__ ref,
                                                   const float* __restrict__ msq,
                                                   const unsigned int* __restrict__ upk,
                                                   float* __restrict__ part,
                                                   int jlen) {
  const int lane = threadIdx.x & 31;
  const int wave = threadIdx.x >> 5;
  const int c    = lane & 15;
  const int hi   = lane >> 4;
  const int i0   = (blockIdx.x * WPB + wave) * 16;
  const int j0   = blockIdx.y * jlen;

  // loop-invariant GEMM1 B operand: (2*log2e)*ref_i^T
  v2f braw = *(const v2f*)(ref + (size_t)(i0 + c) * 4 + 2 * hi);
  v2f bfrag = {braw.x * LOG2E2, braw.y * LOG2E2};

  const char* refB = (const char*)ref;
  const char* upkB = (const char*)upk;
  unsigned aOff = (unsigned)(j0 + c) * 16u + 8u * (unsigned)hi;
  unsigned uOff = (unsigned)(j0 >> 5) * 1024u + (unsigned)lane * 32u;

  v8f acc0 = {}, acc1 = {}, acc2 = {}, acc3 = {};
  const int niter = jlen >> 7;                 // 128 j per iteration
#pragma unroll 2
  for (int it = 0; it < niter; ++it) {
    gauss_chunk(refB, upkB, aOff,        uOff,        bfrag, acc0);
    gauss_chunk(refB, upkB, aOff + 512,  uOff + 1024, bfrag, acc1);
    gauss_chunk(refB, upkB, aOff + 1024, uOff + 2048, bfrag, acc2);
    gauss_chunk(refB, upkB, aOff + 1536, uOff + 3072, bfrag, acc3);
    aOff += 2048;   // 128 rows * 16B
    uOff += 4096;   // 4 tiles * 1KB
  }

  // epilogue: scale output rows by e^{-|r_i|^2}
  v4f m1 = *(const v4f*)(msq + i0 + 8 * hi);
  v4f m2 = *(const v4f*)(msq + i0 + 8 * hi + 4);
  float wi[8] = { __builtin_amdgcn_exp2f(m1.x), __builtin_amdgcn_exp2f(m1.y),
                  __builtin_amdgcn_exp2f(m1.z), __builtin_amdgcn_exp2f(m1.w),
                  __builtin_amdgcn_exp2f(m2.x), __builtin_amdgcn_exp2f(m2.y),
                  __builtin_amdgcn_exp2f(m2.z), __builtin_amdgcn_exp2f(m2.w) };
  v8f acc = (acc0 + acc1) + (acc2 + acc3);
  float* po = part + (size_t)blockIdx.y * NROW * LCH;
#pragma unroll
  for (int v = 0; v < 8; ++v)
    po[(size_t)(i0 + 8 * hi + v) * LCH + c] = acc[v] * wi[v];
}

// ---------------- deterministic reduction over j-splits ----------------
__global__ __launch_bounds__(256) void k_reduce(const float* __restrict__ part,
                                                float* __restrict__ out, int js) {
  int t = blockIdx.x * 256 + threadIdx.x;
  const int NL = NROW * LCH;
  if (t < NL) {
    float s = 0.0f;
    for (int y = 0; y < js; ++y) s += part[t + (size_t)y * NL];
    out[t] = s;
  }
}

extern "C" void kernel_launch(void* const* d_in, const int* in_sizes, int n_in,
                              void* d_out, int out_size, void* d_ws, size_t ws_size,
                              hipStream_t stream) {
  const float* U   = (const float*)d_in[0];   // [8192,16] f32
  const float* ref = (const float*)d_in[1];   // [8192,4]  f32
  float* out = (float*)d_out;                 // [8192,16] f32

  // workspace: msq (32KB) | upk (256KB) | partials (JS * 512KB)
  const size_t msq_bytes = NROW * sizeof(float);
  const size_t upk_bytes = (size_t)(NROW / 32) * 256 * sizeof(unsigned int);
  const size_t base      = msq_bytes + upk_bytes;
  const size_t NLb       = (size_t)NROW * LCH * sizeof(float);

  float*        msq = (float*)d_ws;
  unsigned int* upk = (unsigned int*)((char*)d_ws + msq_bytes);
  float*        prt = (float*)((char*)d_ws + base);

  // adaptive j-split (deterministic: depends only on fixed ws_size)
  int JS = 1;
  if      (ws_size >= base + 8 * NLb) JS = 8;
  else if (ws_size >= base + 4 * NLb) JS = 4;
  else if (ws_size >= base + 2 * NLb) JS = 2;
  float* part = (JS == 1) ? out : prt;

  k_msq<<<NROW / 256, 256, 0, stream>>>(ref, msq);
  k_upk<<<(NROW / 32) * 256 / 256, 256, 0, stream>>>(U, msq, upk);

  dim3 grid(NROW / (WPB * 16), JS);
  k_main<<<grid, WPB * 32, 0, stream>>>(ref, msq, upk, part, NROW / JS);

  if (JS > 1)
    k_reduce<<<(NROW * LCH) / 256, 256, 0, stream>>>(part, out, JS);
}
